// ConcatSquashLinearSA3_69020124446959
// MI455X (gfx1250) — matrix operand order repaired and checked by reading, not compiled, not run on hardware
//
#include <hip/hip_runtime.h>
#include <hip/hip_bf16.h>

typedef __attribute__((ext_vector_type(16))) _Float16 v16h;
typedef __attribute__((ext_vector_type(8)))  _Float16 v8h;
typedef __attribute__((ext_vector_type(8)))  float    v8f;

#define DIM_IN  128
#define DIM_OUT 256
#define DIM_CTX 259
#define BATCH   32
#define NROWS   2048
#define TOT_ROWS (BATCH * NROWS)                 // 65536
#define TOT_ELEMS ((size_t)TOT_ROWS * DIM_OUT)   // 16,777,216

// ---------------------------------------------------------------------------
// WMMA fragment loaders (wave32, V_WMMA_F32_16X16X32_F16 layouts per ISA 7.12.2)
// ---------------------------------------------------------------------------

// A (16x32, MxK) from row-major f32 matrix with leading dim `ld`.
__device__ __forceinline__ v16h load_a_frag_g(const float* __restrict__ src, int ld,
                                              int row0, int k0) {
  int lane = threadIdx.x & 31;
  int r = lane & 15;
  int kb = (lane & 16) ? 8 : 0;
  const float* p = src + (size_t)(row0 + r) * ld + k0 + kb;
  const float4* p4 = (const float4*)p;   // 16B aligned
  float4 q0 = p4[0], q1 = p4[1], q2 = p4[4], q3 = p4[5];
  v16h a;
  a[0]=(_Float16)q0.x;  a[1]=(_Float16)q0.y;  a[2]=(_Float16)q0.z;  a[3]=(_Float16)q0.w;
  a[4]=(_Float16)q1.x;  a[5]=(_Float16)q1.y;  a[6]=(_Float16)q1.z;  a[7]=(_Float16)q1.w;
  a[8]=(_Float16)q2.x;  a[9]=(_Float16)q2.y;  a[10]=(_Float16)q2.z; a[11]=(_Float16)q2.w;
  a[12]=(_Float16)q3.x; a[13]=(_Float16)q3.y; a[14]=(_Float16)q3.z; a[15]=(_Float16)q3.w;
  return a;
}

// A (16x32) from row-major f16 matrix (global or LDS), leading dim in halves.
__device__ __forceinline__ v16h load_a_frag_h(const _Float16* src, int ld,
                                              int row0, int k0) {
  int lane = threadIdx.x & 31;
  int r = lane & 15;
  int kb = (lane & 16) ? 8 : 0;
  const _Float16* p = src + (size_t)(row0 + r) * ld + k0 + kb;  // 16B aligned
  v8h lo = *(const v8h*)p;
  v8h hi = *(const v8h*)(p + 16);
  v16h a;
#pragma unroll
  for (int h = 0; h < 8; ++h) { a[h] = lo[h]; a[h + 8] = hi[h]; }
  return a;
}

// B (32x16, KxN) with B[k][n] = M[n0+n][k0+k], M row-major f16, leading dim ld.
// lanes 0-15: col=lane, K 0..7,16..23 ; lanes 16-31: col=lane-16, K 8..15,24..31 (table order).
__device__ __forceinline__ v16h load_b_frag_h(const _Float16* __restrict__ M, int ld,
                                              int n0, int k0) {
  int lane = threadIdx.x & 31;
  int c = lane & 15;
  int kb = (lane & 16) ? 8 : 0;
  const _Float16* p = M + (size_t)(n0 + c) * ld + k0 + kb;      // 16B aligned
  v8h lo = *(const v8h*)p;
  v8h hi = *(const v8h*)(p + 16);
  v16h b;
#pragma unroll
  for (int h = 0; h < 8; ++h) { b[h] = lo[h]; b[h + 8] = hi[h]; }
  return b;
}

// ---------------------------------------------------------------------------
// K0: f32 -> f16 weight conversion
// ---------------------------------------------------------------------------
__global__ void __launch_bounds__(256) k0_cvt(const float* __restrict__ src,
                                              _Float16* __restrict__ dst, int n) {
  int i = blockIdx.x * 256 + threadIdx.x;
  if (i < n) dst[i] = (_Float16)src[i];
}

// ---------------------------------------------------------------------------
// K1: ctx-side math -> gate[b][o], hbias[b][o], attT[b][e][o] (f16)  (32 blocks)
// ---------------------------------------------------------------------------
__global__ void __launch_bounds__(256) k1_ctx(
    const float* __restrict__ ctx, const float* __restrict__ Wg, const float* __restrict__ bg,
    const float* __restrict__ Wh, const float* __restrict__ Wk, const float* __restrict__ Wv,
    float* __restrict__ gate, float* __restrict__ hbias, _Float16* __restrict__ attTh) {
  int b = blockIdx.x, o = threadIdx.x;
  __shared__ float sc[DIM_CTX];
  __shared__ float kk[256], vv[256], mrow[256], srow[256], cinv[256];
  for (int i = o; i < DIM_CTX; i += 256) sc[i] = ctx[b * DIM_CTX + i];
  __syncthreads();

  float dg = 0.f, dh = 0.f, dk = 0.f, dv = 0.f;
  for (int i = 0; i < DIM_CTX; ++i) {
    float cv = sc[i];
    dg = fmaf(cv, Wg[(size_t)o * DIM_CTX + i], dg);
    dh = fmaf(cv, Wh[(size_t)o * DIM_CTX + i], dh);
  }
  for (int i = 0; i < 256; ++i) {
    float cv = sc[i];
    dk = fmaf(cv, Wk[(size_t)o * 256 + i], dk);
    dv = fmaf(cv, Wv[(size_t)o * 256 + i], dv);
  }
  gate[b * 256 + o]  = 1.0f / (1.0f + __expf(-(dg + bg[o])));
  hbias[b * 256 + o] = dh;
  kk[o] = dk; vv[o] = dv;
  __syncthreads();

  // row-wise softmax stats for row o: energy[o][e] = kk[o]*vv[e]
  float ko = kk[o];
  float m = -3.4e38f;
  for (int e = 0; e < 256; ++e) m = fmaxf(m, ko * vv[e]);
  float s = 0.f;
  for (int e = 0; e < 256; ++e) s += __expf(ko * vv[e] - m);
  mrow[o] = m; srow[o] = s;
  __syncthreads();

  // column sums (thread index plays role of e)
  float ve = vv[o];
  float cs = 0.f;
  for (int r = 0; r < 256; ++r) cs += __expf(kk[r] * ve - mrow[r]) / srow[r];
  cinv[o] = 1.0f / (1e-9f + cs);
  __syncthreads();

  // attT[b][e][o] (f16), coalesced in o
  _Float16* dst = attTh + (size_t)b * 65536;
  float invs = 1.0f / srow[o];
  for (int e = 0; e < 256; ++e) {
    float val = __expf(ko * vv[e] - mrow[o]) * invs * cinv[e];
    dst[(size_t)e * 256 + o] = (_Float16)val;
  }
}

// ---------------------------------------------------------------------------
// K2: x1 = x @ W_layer^T + b_layer ; also emit f16 shadow x1h
//     512 blocks x 256 thr = 8 waves x 16 rows
// ---------------------------------------------------------------------------
__global__ void __launch_bounds__(256) k2_gemm1(
    const float* __restrict__ x, const _Float16* __restrict__ Wlh,
    const float* __restrict__ bl, float* __restrict__ x1, _Float16* __restrict__ x1h) {
  int wave = threadIdx.x >> 5, lane = threadIdx.x & 31;
  int row0 = blockIdx.x * 128 + wave * 16;
  int col  = lane & 15;
  int roff = (lane & 16) ? 8 : 0;

  v16h afr[4];
#pragma unroll
  for (int kt = 0; kt < 4; ++kt) afr[kt] = load_a_frag_g(x, DIM_IN, row0, kt * 32);

  for (int nt = 0; nt < 16; ++nt) {
    v8f c = {};
#pragma unroll
    for (int kt = 0; kt < 4; ++kt) {
      v16h bf = load_b_frag_h(Wlh, DIM_IN, nt * 16, kt * 32);
      c = __builtin_amdgcn_wmma_f32_16x16x32_f16(false, afr[kt], false, bf,
                                                 (short)0, c, false, false);
    }
    float bb = bl[nt * 16 + col];
#pragma unroll
    for (int v = 0; v < 8; ++v) {
      float val = c[v] + bb;
      size_t idx = (size_t)(row0 + roff + v) * DIM_OUT + nt * 16 + col;
      x1[idx]  = val;
      x1h[idx] = (_Float16)val;
    }
  }
}

// ---------------------------------------------------------------------------
// K3: x_r = x1 @ att[b]; u = x1 - x_r (f16, LDS); t = u @ W_tc^T + b_tc
//     1024 blocks x 128 thr (4 waves x 16 rows), LDS 4*16*264*2 = 33 KB
// ---------------------------------------------------------------------------
#define K3_WAVES 4
#define ULD 264
__global__ void __launch_bounds__(128) k3_attn_tc(
    const float* __restrict__ x1g, const _Float16* __restrict__ x1h,
    const _Float16* __restrict__ attTh, const _Float16* __restrict__ Wtch,
    const float* __restrict__ btc, float* __restrict__ tg) {
  __shared__ _Float16 ush[K3_WAVES][16][ULD];
  int wave = threadIdx.x >> 5, lane = threadIdx.x & 31;
  int b    = blockIdx.x >> 5;
  int grp  = blockIdx.x & 31;
  int nrow0 = (grp * K3_WAVES + wave) * 16;              // row within batch
  const float*    x1b  = x1g   + (size_t)b * NROWS * DIM_OUT;
  const _Float16* x1hb = x1h   + (size_t)b * NROWS * DIM_OUT;
  const _Float16* attb = attTh + (size_t)b * 256 * 256;
  float*          tb   = tg    + (size_t)b * NROWS * DIM_OUT;
  int col  = lane & 15;
  int roff = (lane & 16) ? 8 : 0;
  _Float16* u = &ush[wave][0][0];

  // phase 1: x_r tiles + u = x1 - x_r into LDS (f16)
  v16h afr[8];
#pragma unroll
  for (int kt = 0; kt < 8; ++kt) afr[kt] = load_a_frag_h(x1hb, DIM_OUT, nrow0, kt * 32);

  for (int et = 0; et < 16; ++et) {
    v8f c = {};
#pragma unroll
    for (int kt = 0; kt < 8; ++kt) {
      v16h bf = load_b_frag_h(attb, 256, et * 16, kt * 32);  // B[k=o][n=e]=att[b][o][e]
      c = __builtin_amdgcn_wmma_f32_16x16x32_f16(false, afr[kt], false, bf,
                                                 (short)0, c, false, false);
    }
#pragma unroll
    for (int v = 0; v < 8; ++v) {
      int mrow = roff + v;
      float xv = x1b[(size_t)(nrow0 + mrow) * DIM_OUT + et * 16 + col];
      u[mrow * ULD + et * 16 + col] = (_Float16)(xv - c[v]);
    }
  }
  __syncthreads();

  // phase 2: t = u @ W_tc^T + b_tc
  v16h ufr[8];
#pragma unroll
  for (int kt = 0; kt < 8; ++kt) ufr[kt] = load_a_frag_h(u, ULD, 0, kt * 32);

  for (int et = 0; et < 16; ++et) {
    v8f c = {};
#pragma unroll
    for (int kt = 0; kt < 8; ++kt) {
      v16h bf = load_b_frag_h(Wtch, 256, et * 16, kt * 32);  // B[k=o][n=e]=Wtc[e][o]
      c = __builtin_amdgcn_wmma_f32_16x16x32_f16(false, ufr[kt], false, bf,
                                                 (short)0, c, false, false);
    }
    float bb = btc[et * 16 + col];
#pragma unroll
    for (int v = 0; v < 8; ++v)
      tb[(size_t)(nrow0 + roff + v) * DIM_OUT + et * 16 + col] = c[v] + bb;
  }
}

// ---------------------------------------------------------------------------
// K4: per-n mean / rstd over (b, e) : 2048 blocks x 256 thr
// ---------------------------------------------------------------------------
__global__ void __launch_bounds__(256) k4_stats(
    const float* __restrict__ t, float* __restrict__ meanv, float* __restrict__ rstd) {
  int n = blockIdx.x, e = threadIdx.x;
  float s = 0.f, ss = 0.f;
#pragma unroll 4
  for (int b = 0; b < BATCH; ++b) {
    float v = t[((size_t)b * NROWS + n) * DIM_OUT + e];
    s += v; ss += v * v;
  }
  __shared__ float rs[256], rss[256];
  rs[e] = s; rss[e] = ss;
  __syncthreads();
  for (int st = 128; st > 0; st >>= 1) {
    if (e < st) { rs[e] += rs[e + st]; rss[e] += rss[e + st]; }
    __syncthreads();
  }
  if (e == 0) {
    float m = rs[0] * (1.0f / 8192.0f);
    float var = rss[0] * (1.0f / 8192.0f) - m * m;
    meanv[n] = m;
    rstd[n] = rsqrtf(var + 1e-5f);
  }
}

// ---------------------------------------------------------------------------
// K5: out = (x1 + relu(bn(t))) * gate + hbias   : 65536 blocks x 256 thr
// ---------------------------------------------------------------------------
__global__ void __launch_bounds__(256) k5_final(
    const float* __restrict__ x1, const float* __restrict__ t,
    const float* __restrict__ meanv, const float* __restrict__ rstd,
    const float* __restrict__ gamma, const float* __restrict__ beta,
    const float* __restrict__ gate, const float* __restrict__ hbias,
    float* __restrict__ out) {
  size_t idx = (size_t)blockIdx.x * 256 + threadIdx.x;
  int e = (int)(idx & 255);
  int n = (int)((idx >> 8) & 2047);
  int b = (int)(idx >> 19);
  float tn = (t[idx] - meanv[n]) * rstd[n] * gamma[n] + beta[n];
  float r = tn > 0.f ? tn : 0.f;
  float xo = x1[idx] + r;
  out[idx] = xo * gate[b * 256 + e] + hbias[b * 256 + e];
}

// ---------------------------------------------------------------------------
extern "C" void kernel_launch(void* const* d_in, const int* in_sizes, int n_in,
                              void* d_out, int out_size, void* d_ws, size_t ws_size,
                              hipStream_t stream) {
  (void)in_sizes; (void)n_in; (void)out_size; (void)ws_size;
  const float* ctx   = (const float*)d_in[0];
  const float* x     = (const float*)d_in[1];
  const float* Wl    = (const float*)d_in[2];
  const float* bl    = (const float*)d_in[3];
  const float* Wh    = (const float*)d_in[4];
  const float* Wg    = (const float*)d_in[5];
  const float* bg    = (const float*)d_in[6];
  const float* Wk    = (const float*)d_in[7];
  const float* Wv    = (const float*)d_in[8];
  const float* Wtc   = (const float*)d_in[9];
  const float* btc   = (const float*)d_in[10];
  const float* gamma = (const float*)d_in[11];
  const float* beta  = (const float*)d_in[12];
  float* out = (float*)d_out;

  // f32 workspace region
  float* ws   = (float*)d_ws;
  float* x1   = ws;                       // 16,777,216 f32
  float* t    = x1 + TOT_ELEMS;           // 16,777,216 f32
  float* gate = t + TOT_ELEMS;            // 8192
  float* hb   = gate + BATCH * 256;       // 8192
  float* mn   = hb + BATCH * 256;         // 2048
  float* rstd = mn + NROWS;               // 2048
  // f16 workspace region (16B-aligned offsets)
  _Float16* x1h   = (_Float16*)(rstd + NROWS);
  _Float16* attTh = x1h + TOT_ELEMS;                  // 2,097,152 halves
  _Float16* Wlh   = attTh + (size_t)BATCH * 256 * 256; // 32,768 halves
  _Float16* Wtch  = Wlh + DIM_OUT * DIM_IN;            // 65,536 halves

  k0_cvt<<<(DIM_OUT * DIM_IN + 255) / 256, 256, 0, stream>>>(Wl, Wlh, DIM_OUT * DIM_IN);
  k0_cvt<<<(DIM_OUT * DIM_OUT + 255) / 256, 256, 0, stream>>>(Wtc, Wtch, DIM_OUT * DIM_OUT);
  k1_ctx<<<BATCH, 256, 0, stream>>>(ctx, Wg, bg, Wh, Wk, Wv, gate, hb, attTh);
  k2_gemm1<<<TOT_ROWS / 128, 256, 0, stream>>>(x, Wlh, bl, x1, x1h);
  k3_attn_tc<<<BATCH * (NROWS / (16 * K3_WAVES)), 128, 0, stream>>>(x1, x1h, attTh, Wtch,
                                                                   btc, t);
  k4_stats<<<NROWS, 256, 0, stream>>>(t, mn, rstd);
  k5_final<<<(int)(TOT_ELEMS / 256), 256, 0, stream>>>(x1, t, mn, rstd, gamma, beta,
                                                       gate, hb, out);
}
